// MessagePassing_52450140618855
// MI455X (gfx1250) — matrix operand order repaired and checked by reading, not compiled, not run on hardware
//
#include <hip/hip_runtime.h>
#include <stdint.h>

// ---------- types ----------
typedef __attribute__((ext_vector_type(16))) __bf16    v16bf;
typedef __attribute__((ext_vector_type(8)))  float     v8f;
typedef __attribute__((ext_vector_type(4)))  unsigned  u32x4;
typedef __attribute__((ext_vector_type(2)))  unsigned  u32x2;

union FragAB { u32x4 u[2]; v16bf v; };   // 32 bytes = 16 bf16 = one WMMA A/B fragment per lane

// ---------- bf16 helpers (manual, RNE) ----------
__device__ __forceinline__ float bf2f(unsigned short b) {
    union { unsigned u; float f; } x; x.u = ((unsigned)b) << 16; return x.f;
}
__device__ __forceinline__ unsigned short f2bf(float f) {
    union { float f; unsigned u; } x; x.f = f;
    unsigned u = x.u;
    unsigned r = u + 0x7FFFu + ((u >> 16) & 1u);   // round-to-nearest-even
    return (unsigned short)(r >> 16);
}

// ---------- small utility kernels ----------
__global__ void k_zero_f32(float* __restrict__ p, int n) {
    int i = blockIdx.x * blockDim.x + threadIdx.x;
    if (i < n) p[i] = 0.0f;
}

__global__ void k_degree(const int* __restrict__ edges, float* __restrict__ deg, int E) {
    int e = blockIdx.x * blockDim.x + threadIdx.x;
    if (e < E) atomicAdd(&deg[edges[2 * e + 1]], 1.0f);
}

__global__ void k_invnorm(const int* __restrict__ edges, const float* __restrict__ deg,
                          float* __restrict__ inv, int E) {
    int e = blockIdx.x * blockDim.x + threadIdx.x;
    if (e < E) {
        int s = edges[2 * e], d = edges[2 * e + 1];
        inv[e] = rsqrtf((deg[d] + 1.0f) * (deg[s] + 1.0f));
    }
}

__global__ void k_cvt_bf16(const float* __restrict__ in, unsigned short* __restrict__ out, int n) {
    int i = blockIdx.x * blockDim.x + threadIdx.x;
    if (i < n) out[i] = f2bf(in[i]);
}

// W is K x N row-major fp32 -> Wt is N x K row-major bf16 (transposed)
__global__ void k_wT_bf16(const float* __restrict__ W, unsigned short* __restrict__ Wt, int K, int N) {
    int i = blockIdx.x * blockDim.x + threadIdx.x;
    if (i < K * N) {
        int k = i / N, n = i % N;
        Wt[(size_t)n * K + k] = f2bf(W[i]);
    }
}

// per edge: agg[dst] += bf2f(h[src]) * inv[e]; 128 lanes per edge, 4 feats/lane (D=512)
__global__ __launch_bounds__(256) void k_scatter(const int* __restrict__ edges,
                                                 const unsigned short* __restrict__ h,
                                                 const float* __restrict__ inv,
                                                 float* __restrict__ agg, int E) {
    int tid = threadIdx.x;
    int e = blockIdx.x * 2 + (tid >> 7);
    if (e >= E) return;
    int li = tid & 127;
    int src = edges[2 * e], dst = edges[2 * e + 1];
    float s = inv[e];
    int d0 = li * 4;
    u32x2 hv = *(const u32x2*)(h + (size_t)src * 512 + d0);
    float* dp = agg + (size_t)dst * 512 + d0;
    atomicAdd(dp + 0, bf2f((unsigned short)(hv.x)) * s);
    atomicAdd(dp + 1, bf2f((unsigned short)(hv.x >> 16)) * s);
    atomicAdd(dp + 2, bf2f((unsigned short)(hv.y)) * s);
    atomicAdd(dp + 3, bf2f((unsigned short)(hv.y >> 16)) * s);
}

// ---------- WMMA K-loop (ISA 16-bit A 16x32 / B 32x16 layouts, wave32) ----------
// A fragment: lanes 0-15 -> row M=lane,    K chunks {k0..k0+7} and {k0+16..k0+23}
//             lanes16-31 -> row M=lane-16, K chunks {k0+8..k0+15} and {k0+24..k0+31}
// B fragment (from B^T, N x K row-major): lane -> column n = n0 + (lane&15),
//             K chunks {k0+8h..k0+8h+7} and {k0+16+8h..k0+16+8h+7}, h = lane>=16
//
// Out-of-range M rows are CLAMPED to M-1 (duplicate valid data). This keeps the
// K-loop branch-free; garbage rows of D are dropped by the store-side guard
// (WMMA output row m depends only on input row m of A, so in-range rows are exact).
__device__ __forceinline__ void gemm_kloop(const unsigned short* __restrict__ A,
                                           const unsigned short* __restrict__ Bt,
                                           int K, int M, int m0, int n0, int lane,
                                           v8f acc[2][4]) {
    const int half = lane >> 4;
    const int r15  = lane & 15;
    int rowA0 = m0 + r15;      if (rowA0 >= M) rowA0 = M - 1;
    int rowA1 = m0 + 16 + r15; if (rowA1 >= M) rowA1 = M - 1;
    const unsigned short* pa0 = A + (size_t)rowA0 * K + half * 8;
    const unsigned short* pa1 = A + (size_t)rowA1 * K + half * 8;
    const unsigned short* pb0 = Bt + (size_t)(n0 +  0 + r15) * K + half * 8;
    const unsigned short* pb1 = Bt + (size_t)(n0 + 16 + r15) * K + half * 8;
    const unsigned short* pb2 = Bt + (size_t)(n0 + 32 + r15) * K + half * 8;
    const unsigned short* pb3 = Bt + (size_t)(n0 + 48 + r15) * K + half * 8;

#pragma unroll 2
    for (int k0 = 0; k0 < K; k0 += 32) {
        FragAB fa0, fa1, fb0, fb1, fb2, fb3;
        fa0.u[0] = *(const u32x4*)(pa0 + k0);
        fa0.u[1] = *(const u32x4*)(pa0 + k0 + 16);
        fa1.u[0] = *(const u32x4*)(pa1 + k0);
        fa1.u[1] = *(const u32x4*)(pa1 + k0 + 16);
        fb0.u[0] = *(const u32x4*)(pb0 + k0);
        fb0.u[1] = *(const u32x4*)(pb0 + k0 + 16);
        fb1.u[0] = *(const u32x4*)(pb1 + k0);
        fb1.u[1] = *(const u32x4*)(pb1 + k0 + 16);
        fb2.u[0] = *(const u32x4*)(pb2 + k0);
        fb2.u[1] = *(const u32x4*)(pb2 + k0 + 16);
        fb3.u[0] = *(const u32x4*)(pb3 + k0);
        fb3.u[1] = *(const u32x4*)(pb3 + k0 + 16);

        acc[0][0] = __builtin_amdgcn_wmma_f32_16x16x32_bf16(false, fa0.v, false, fb0.v, (short)0, acc[0][0], false, false);
        acc[0][1] = __builtin_amdgcn_wmma_f32_16x16x32_bf16(false, fa0.v, false, fb1.v, (short)0, acc[0][1], false, false);
        acc[0][2] = __builtin_amdgcn_wmma_f32_16x16x32_bf16(false, fa0.v, false, fb2.v, (short)0, acc[0][2], false, false);
        acc[0][3] = __builtin_amdgcn_wmma_f32_16x16x32_bf16(false, fa0.v, false, fb3.v, (short)0, acc[0][3], false, false);
        acc[1][0] = __builtin_amdgcn_wmma_f32_16x16x32_bf16(false, fa1.v, false, fb0.v, (short)0, acc[1][0], false, false);
        acc[1][1] = __builtin_amdgcn_wmma_f32_16x16x32_bf16(false, fa1.v, false, fb1.v, (short)0, acc[1][1], false, false);
        acc[1][2] = __builtin_amdgcn_wmma_f32_16x16x32_bf16(false, fa1.v, false, fb2.v, (short)0, acc[1][2], false, false);
        acc[1][3] = __builtin_amdgcn_wmma_f32_16x16x32_bf16(false, fa1.v, false, fb3.v, (short)0, acc[1][3], false, false);
    }
}

// C = act(A1@B1 + A2@B2 + bias); block tile 128x128, 8 waves, wave tile 32x64
__global__ __launch_bounds__(256) void k_gemm_bf16(
    const unsigned short* __restrict__ A1, const unsigned short* __restrict__ B1t, int K1,
    const unsigned short* __restrict__ A2, const unsigned short* __restrict__ B2t, int K2,
    const float* __restrict__ bias,
    unsigned short* __restrict__ Cb, float* __restrict__ Cf,
    int M, int N, int do_relu) {
    int lane = threadIdx.x & 31;
    int w = threadIdx.x >> 5;
    int wm = w & 3, wn = w >> 2;
    int m0 = blockIdx.x * 128 + wm * 32;
    int n0 = blockIdx.y * 128 + wn * 64;

    v8f acc[2][4] = {};
    gemm_kloop(A1, B1t, K1, M, m0, n0, lane, acc);
    if (A2) gemm_kloop(A2, B2t, K2, M, m0, n0, lane, acc);

    // C/D layout: VGPR r, lane<16 -> row m0+i*16+r, lane>=16 -> row m0+i*16+8+r; col n0+j*16+(lane&15)
    int rofs = (lane >> 4) * 8;
    int cofs = lane & 15;
#pragma unroll
    for (int i = 0; i < 2; ++i) {
#pragma unroll
        for (int j = 0; j < 4; ++j) {
            int col = n0 + j * 16 + cofs;
            float bv = bias ? bias[col] : 0.0f;
#pragma unroll
            for (int r = 0; r < 8; ++r) {
                int row = m0 + i * 16 + rofs + r;
                if (row < M) {
                    float vv = acc[i][j][r] + bv;
                    if (do_relu) vv = vv > 0.0f ? vv : 0.0f;
                    if (Cb) Cb[(size_t)row * N + col] = f2bf(vv);
                    if (Cf) Cf[(size_t)row * N + col] = vv;
                }
            }
        }
    }
}

// ---------- host-side launch ----------
extern "C" void kernel_launch(void* const* d_in, const int* in_sizes, int n_in,
                              void* d_out, int out_size, void* d_ws, size_t ws_size,
                              hipStream_t stream) {
    const int N = 20000, E = 160000, DLAT = 512, DOUT = 256, L = 3;
    const size_t ND = (size_t)N * DLAT;

    const float* features = (const float*)d_in[0];
    const int*   edges    = (const int*)d_in[1];
    const float* W_enc    = (const float*)d_in[2];
    const float* b_enc    = (const float*)d_in[3];
    const float* W_self   = (const float*)d_in[4];
    const float* W_neigh  = (const float*)d_in[5];
    const float* b_comb   = (const float*)d_in[6];
    const float* W_out    = (const float*)d_in[7];
    const float* b_out    = (const float*)d_in[8];

    char* ws = (char*)d_ws;
    size_t off = 0;
    auto alloc = [&](size_t bytes) -> char* {
        char* p = ws + off;
        off += (bytes + 255) & ~(size_t)255;
        return p;
    };
    float*          deg    = (float*)alloc((size_t)N * 4);
    float*          inv    = (float*)alloc((size_t)E * 4);
    unsigned short* featb  = (unsigned short*)alloc(ND * 2);
    unsigned short* hb0    = (unsigned short*)alloc(ND * 2);
    unsigned short* hb1    = (unsigned short*)alloc(ND * 2);
    float*          aggf   = (float*)alloc(ND * 4);
    unsigned short* aggb   = (unsigned short*)alloc(ND * 2);
    unsigned short* WencT  = (unsigned short*)alloc((size_t)DLAT * DLAT * 2);
    unsigned short* WselfT = (unsigned short*)alloc((size_t)L * DLAT * DLAT * 2);
    unsigned short* WneighT= (unsigned short*)alloc((size_t)L * DLAT * DLAT * 2);
    unsigned short* WoutT  = (unsigned short*)alloc((size_t)DLAT * DOUT * 2);

    const int T = 256;
    // degrees + norms
    k_zero_f32<<<(N + T - 1) / T, T, 0, stream>>>(deg, N);
    k_degree<<<(E + T - 1) / T, T, 0, stream>>>(edges, deg, E);
    k_invnorm<<<(E + T - 1) / T, T, 0, stream>>>(edges, deg, inv, E);

    // conversions
    k_cvt_bf16<<<((int)ND + T - 1) / T, T, 0, stream>>>(features, featb, (int)ND);
    k_wT_bf16<<<(DLAT * DLAT + T - 1) / T, T, 0, stream>>>(W_enc, WencT, DLAT, DLAT);
    for (int l = 0; l < L; ++l) {
        k_wT_bf16<<<(DLAT * DLAT + T - 1) / T, T, 0, stream>>>(
            W_self + (size_t)l * DLAT * DLAT, WselfT + (size_t)l * DLAT * DLAT, DLAT, DLAT);
        k_wT_bf16<<<(DLAT * DLAT + T - 1) / T, T, 0, stream>>>(
            W_neigh + (size_t)l * DLAT * DLAT, WneighT + (size_t)l * DLAT * DLAT, DLAT, DLAT);
    }
    k_wT_bf16<<<(DLAT * DOUT + T - 1) / T, T, 0, stream>>>(W_out, WoutT, DLAT, DOUT);

    // encoder: h = features @ W_enc + b_enc
    {
        dim3 g((N + 127) / 128, DLAT / 128);
        k_gemm_bf16<<<g, T, 0, stream>>>(featb, WencT, DLAT,
                                         (const unsigned short*)nullptr, (const unsigned short*)nullptr, 0,
                                         b_enc, hb0, (float*)nullptr, N, DLAT, 0);
    }

    unsigned short* hcur = hb0;
    unsigned short* hnxt = hb1;
    for (int l = 0; l < L; ++l) {
        k_zero_f32<<<((int)ND + T - 1) / T, T, 0, stream>>>(aggf, (int)ND);
        k_scatter<<<(E + 1) / 2, T, 0, stream>>>(edges, hcur, inv, aggf, E);
        k_cvt_bf16<<<((int)ND + T - 1) / T, T, 0, stream>>>(aggf, aggb, (int)ND);
        dim3 g((N + 127) / 128, DLAT / 128);
        k_gemm_bf16<<<g, T, 0, stream>>>(hcur, WselfT + (size_t)l * DLAT * DLAT, DLAT,
                                         aggb, WneighT + (size_t)l * DLAT * DLAT, DLAT,
                                         b_comb + (size_t)l * DLAT, hnxt, (float*)nullptr,
                                         N, DLAT, 1);
        unsigned short* t = hcur; hcur = hnxt; hnxt = t;
    }

    // out = h @ W_out + b_out  (fp32 output)
    {
        dim3 g((N + 127) / 128, DOUT / 128);
        k_gemm_bf16<<<g, T, 0, stream>>>(hcur, WoutT, DLAT,
                                         (const unsigned short*)nullptr, (const unsigned short*)nullptr, 0,
                                         b_out, (unsigned short*)nullptr, (float*)d_out,
                                         N, DOUT, 0);
    }
}
